// PointCloud3DCNN_55396488184472
// MI455X (gfx1250) — matrix operand order repaired and checked by reading, not compiled, not run on hardware
//
#include <hip/hip_runtime.h>
#include <hip/hip_bf16.h>
#include <math.h>

// ---------------------------------------------------------------------------
// Masked 3D U-Net for MI455X (gfx1250, wave32).
// Convs/deconvs = implicit GEMM on v_wmma_f32_16x16x32_bf16.
//  - A pre-swizzled into exact WMMA fragment order -> 32B contiguous per lane
//  - bf16 activations channel-last (NDHWC) with a 32B zero guard page at the
//    base: out-of-bounds taps load index -16 -> zeros, no post-load masking
//  - dual 16x16 N-tiles per wave: one A fragment + tap feeds two v_wmma
//  - layer-1 (Cin padded 3->4): fully constexpr taps, zero divisions
// ---------------------------------------------------------------------------

typedef __attribute__((ext_vector_type(16))) __bf16    v16bf;
typedef __attribute__((ext_vector_type(8)))  float     v8f;
typedef __attribute__((ext_vector_type(8)))  unsigned  v8u;
typedef __attribute__((ext_vector_type(2)))  unsigned  v2u;

#define BN_EPS 1e-5f
#define ZPAD   16            // guard elements (32B) at base of bf16 buffers

template <int V> struct IC { static constexpr int value = V; };

__device__ __forceinline__ __bf16 f2bf(float f) {
  unsigned u = __builtin_bit_cast(unsigned, f);
  u += 0x7FFFu + ((u >> 16) & 1u);
  unsigned short h = (unsigned short)(u >> 16);
  return __builtin_bit_cast(__bf16, h);
}

__device__ __forceinline__ float sigmoidf(float x) {
  return 1.0f / (1.0f + __expf(-x));
}

// spatial tap: output voxel (oz,oy,ox) + kernel element kel -> input voxel
template <int DIN, int MODE>
__device__ __forceinline__ bool tap(int kel, int oz, int oy, int ox, int& spin) {
  int kd = kel / 9;
  int r  = kel - kd * 9;
  int kh = r / 3;
  int kw = r - kh * 3;
  int iz, iy, ix; bool ok;
  if (MODE == 0) {                       // stride-2 conv, pad 1
    iz = 2 * oz - 1 + kd; iy = 2 * oy - 1 + kh; ix = 2 * ox - 1 + kw;
    ok = ((unsigned)iz < (unsigned)DIN) & ((unsigned)iy < (unsigned)DIN) &
         ((unsigned)ix < (unsigned)DIN);
  } else {                               // lhs_dilation 2, pad (1,2), W pre-flipped
    int tz = oz + kd - 1, ty = oy + kh - 1, tx = ox + kw - 1;
    iz = tz >> 1; iy = ty >> 1; ix = tx >> 1;
    ok = !((tz | ty | tx) & 1) &&
         ((unsigned)iz < (unsigned)DIN) & ((unsigned)iy < (unsigned)DIN) &
         ((unsigned)ix < (unsigned)DIN);
  }
  spin = (iz * DIN + iy) * DIN + ix;
  return ok;
}

// ---------------------------------------------------------------------------
// A-pack: swizzle bf16 weights into per-(kt32, mtile) WMMA A-fragment order.
// K-index k = kel*CINP + ci (CINP possibly padded); valid iff ci<CINW, kel<27.
// ---------------------------------------------------------------------------
template <int CINP, int CINW, int COUT, int MODE>
__global__ void k_pack_a(const float* __restrict__ W, __bf16* __restrict__ Aswz,
                         int n) {
  constexpr int COUTP = (COUT + 15) & ~15;
  constexpr int MT = COUTP / 16;
  int i = blockIdx.x * blockDim.x + threadIdx.x;
  if (i >= n) return;
  int e = i & 15;
  int lane = (i >> 4) & 31;
  int rest = i >> 9;                 // kt32*MT + mtile
  int mtile = rest % MT;
  int kt32 = rest / MT;
  int lhalf = (lane < 16) ? 0 : 16;
  int ka = kt32 * 32 + lhalf + e;
  int m  = mtile * 16 + (lane & 15);
  int kel = ka / CINP, ci = ka - kel * CINP;
  float v = 0.0f;
  if (kel < 27 && ci < CINW && m < COUT) {
    int keli = kel;
    if (MODE != 0) {
      int kd = kel / 9, r2 = kel - kd * 9, kh = r2 / 3, kw = r2 - kh * 3;
      keli = (2 - kd) * 9 + (2 - kh) * 3 + (2 - kw);       // spatial flip
    }
    v = W[(keli * CINW + ci) * COUT + m];
  }
  Aswz[i] = f2bf(v);
}

// ---------------------------------------------------------------------------
// Implicit-GEMM conv/deconv.
//  CIN>=16: dual 16x16 N-tiles per wave (A fragment reused across 2 wmma)
//  CIN==4 : single tile, constexpr K-steps (zero-division taps)
// Xb points ZPAD elements past a 32B zero page; invalid taps use index -ZPAD.
// ---------------------------------------------------------------------------
template <int CIN, int COUT, int DIN, int MODE>
__global__ void __launch_bounds__(256)
k_gemm_wmma(const __bf16* __restrict__ Xb, const __bf16* __restrict__ Aswz,
            float* __restrict__ Y) {
  constexpr int DOUT  = (MODE == 0) ? (DIN / 2) : (DIN * 2);
  constexpr int D2    = DOUT * DOUT;
  constexpr int D3    = D2 * DOUT;
  constexpr int DIN3  = DIN * DIN * DIN;
  constexpr int NVOX  = 2 * D3;                 // B = 2
  constexpr int NTILES = NVOX / 16;
  constexpr int K     = CIN * 27;
  constexpr int KP    = (K + 31) & ~31;
  constexpr int COUTP = (COUT + 15) & ~15;
  constexpr int MT    = COUTP / 16;

  const int lane  = threadIdx.x & 31;
  const int wave  = threadIdx.x >> 5;
  const int mtile = blockIdx.y;
  const int l15   = lane & 15;
  const int lhalf = (lane < 16) ? 0 : 8;

  // ncol -> (b, oz, oy, ox); B=2 so batch is a compare, not a division
  auto dec = [&](int ncol, int& b, int& oz, int& oy, int& ox) {
    b = (ncol >= D3) ? 1 : 0;
    int r = ncol - b * D3;
    oz = r / D2;  r -= oz * D2;
    oy = r / DOUT; ox = r - oy * DOUT;
  };
  auto aload = [&](int kt) {
    const __bf16* ap = Aswz + (size_t)((kt >> 5) * MT + mtile) * 512 + lane * 16;
    if (kt + 32 < KP) __builtin_prefetch(ap + MT * 512, 0, 1);
    return *(const v16bf*)ap;                   // 2 x global_load_b128
  };
  auto store = [&](const v8f& a, int bb, int nsp) {
#pragma unroll
    for (int r = 0; r < 8; ++r) {
      int mout = mtile * 16 + lhalf + r;
      if (mout < COUT) Y[(bb * COUT + mout) * D3 + nsp] = a[r];
    }
  };

  if constexpr (CIN == 4) {
    // ---- layer 1: single tile, 4 constexpr K-steps ----
    const int ntile = blockIdx.x * 8 + wave;    // wave-uniform
    if (ntile >= NTILES) return;
    int b, oz, oy, ox;
    dec(ntile * 16 + l15, b, oz, oy, ox);
    const bool lo = (lane < 16);
    v8f acc = {};
    auto step = [&](auto kc) {
      constexpr int kt = decltype(kc)::value;
      v16bf amat = aload(kt);
      v8u bw;
#pragma unroll
      for (int g = 0; g < 4; ++g) {             // both lane-half candidates are
        const int kb0 = kt / 4 + g;             // literal -> taps fold to consts
        const int kb1 = kb0 + 4;
        int spin0, spin1;
        bool ok0 = tap<DIN, MODE>(kb0 > 26 ? 26 : kb0, oz, oy, ox, spin0) &&
                   (kb0 < 27);
        bool ok1 = tap<DIN, MODE>(kb1 > 26 ? 26 : kb1, oz, oy, ox, spin1) &&
                   (kb1 < 27);
        int idx0 = ok0 ? (b * DIN3 + spin0) * 4 : -ZPAD;   // zero guard page
        int idx1 = ok1 ? (b * DIN3 + spin1) * 4 : -ZPAD;
        int bidx = lo ? idx0 : idx1;
        v2u t = *(const v2u*)(Xb + bidx);       // global_load_b64, no masking
        bw[2 * g]     = t[0];
        bw[2 * g + 1] = t[1];
      }
      v16bf bmat = __builtin_bit_cast(v16bf, bw);
      acc = __builtin_amdgcn_wmma_f32_16x16x32_bf16(false, amat, false, bmat,
                                                    (short)0, acc, false, false);
    };
    step(IC<0>{}); step(IC<32>{}); step(IC<64>{}); step(IC<96>{});
    store(acc, b, (oz * DOUT + oy) * DOUT + ox);
  } else {
    // ---- dual N-tiles per wave ----
    constexpr int NT2 = (NTILES + 1) / 2;
    const int pair = blockIdx.x * 8 + wave;     // wave-uniform
    if (pair >= NT2) return;
    const int ntile0 = 2 * pair;
    const bool has1 = (ntile0 + 1) < NTILES;    // wave-uniform
    int b0, oz0, oy0, ox0, b1, oz1, oy1, ox1;
    dec(ntile0 * 16 + l15, b0, oz0, oy0, ox0);
    dec(ntile0 * 16 + 16 + l15, b1, oz1, oy1, ox1);

    v8f acc0 = {}, acc1 = {};
    auto bload = [&](int kel, int cih, int b, int oz, int oy, int ox) {
      int spin;
      bool ok = tap<DIN, MODE>(kel, oz, oy, ox, spin);
      int bidx = (b * DIN3 + spin) * CIN + cih; // 32B-aligned, contiguous
      bidx = ok ? bidx : -ZPAD;                 // zero guard page
      return *(const v16bf*)(Xb + bidx);        // 2 x global_load_b128, pure
    };

#pragma unroll 2
    for (int kt = 0; kt < KP; kt += 32) {
      v16bf amat = aload(kt);
      int kel, cih;
      if constexpr (CIN >= 32) {                // whole tile inside one tap
        kel = kt / CIN;
        cih = (kt % CIN) + ((lane < 16) ? 0 : 16);
      } else {                                  // CIN==16: one tap per lane-half
        kel = kt / 16 + ((lane < 16) ? 0 : 1);
        cih = 0;
      }
      v16bf bm0 = bload(kel, cih, b0, oz0, oy0, ox0);
      acc0 = __builtin_amdgcn_wmma_f32_16x16x32_bf16(false, amat, false, bm0,
                                                     (short)0, acc0, false, false);
      if (has1) {
        v16bf bm1 = bload(kel, cih, b1, oz1, oy1, ox1);
        acc1 = __builtin_amdgcn_wmma_f32_16x16x32_bf16(false, amat, false, bm1,
                                                       (short)0, acc1, false, false);
      }
    }
    store(acc0, b0, (oz0 * DOUT + oy0) * DOUT + ox0);
    if (has1) store(acc1, b1, (oz1 * DOUT + oy1) * DOUT + ox1);
  }
}

// ---------------------------------------------------------------------------
// Elementwise / reduction kernels (templated; batch index via compare, B=2)
// f32 tensors NCDHW; bf16 copies NDHWC.
// ---------------------------------------------------------------------------
__global__ void k_mask_cast(const unsigned char* __restrict__ mb,
                            float* __restrict__ m0, int n) {
  int i = blockIdx.x * blockDim.x + threadIdx.x;
  if (i < n) m0[i] = mb[i] ? 1.0f : 0.0f;
}

// masked input -> bf16 NDHWC with C padded 3->4 (channel 3 = 0)
template <int D3>
__global__ void k_apply_mask_c4(const float* __restrict__ f,
                                const float* __restrict__ m,
                                __bf16* __restrict__ xb, int n) {
  int i = blockIdx.x * blockDim.x + threadIdx.x;   // i = b*D3 + sp
  if (i >= n) return;
  float mv = m[i];
  int b = (i >= D3) ? 1 : 0;
  int sp = i - b * D3;
  const float* fb = f + (size_t)b * 3 * D3 + sp;
  __bf16* o = xb + (size_t)i * 4;
  o[0] = f2bf(fb[0] * mv);
  o[1] = f2bf(fb[(size_t)D3] * mv);
  o[2] = f2bf(fb[(size_t)2 * D3] * mv);
  o[3] = __builtin_bit_cast(__bf16, (unsigned short)0);
}

template <int Do>
__global__ void k_pool2(const float* __restrict__ mi, float* __restrict__ mo,
                        int n) {
  constexpr int Do2 = Do * Do, Do3 = Do2 * Do;
  constexpr int Di = Do * 2, Di2 = Di * Di, Di3 = Di2 * Di;
  int i = blockIdx.x * blockDim.x + threadIdx.x;
  if (i >= n) return;
  int b = (i >= Do3) ? 1 : 0;
  int r = i - b * Do3;
  int z = r / Do2; r -= z * Do2;
  int y = r / Do;  int x = r - y * Do;
  const float* base = mi + b * Di3;
  float mx = 0.0f;
  for (int dz = 0; dz < 2; ++dz)
    for (int dy = 0; dy < 2; ++dy)
      for (int dx = 0; dx < 2; ++dx)
        mx = fmaxf(mx, base[(2 * z + dz) * Di2 + (2 * y + dy) * Di + (2 * x + dx)]);
  mo[i] = mx;
}

// stats: [0]=cnt, [1..C]=sum(y*m), [1+C..2C]=sum((y-mean)^2*m)
template <int C, int D3>
__global__ void __launch_bounds__(256)
k_bn_stats(const float* __restrict__ y, const float* __restrict__ m,
           float* __restrict__ stats, int pass) {
  __shared__ float sv[256];
  __shared__ float sc[256];
  const int c = blockIdx.x;
  constexpr int nvox = 2 * D3;
  float mean = 0.0f;
  if (pass == 2) mean = stats[1 + c] / stats[0];
  float lv = 0.0f, lc = 0.0f;
  for (int v = blockIdx.y * blockDim.x + threadIdx.x; v < nvox;
       v += blockDim.x * gridDim.y) {
    int b = (v >= D3) ? 1 : 0;
    int sp = v - b * D3;
    float mv = m[v];
    float yv = y[(b * C + c) * D3 + sp];
    if (pass == 1) { lv += yv * mv; lc += mv; }
    else           { float d = yv - mean; lv += d * d * mv; }
  }
  sv[threadIdx.x] = lv; sc[threadIdx.x] = lc;
  __syncthreads();
  for (int s = 128; s > 0; s >>= 1) {
    if ((int)threadIdx.x < s) {
      sv[threadIdx.x] += sv[threadIdx.x + s];
      sc[threadIdx.x] += sc[threadIdx.x + s];
    }
    __syncthreads();
  }
  if (threadIdx.x == 0) {
    atomicAdd(&stats[(pass == 1 ? 1 : 1 + C) + c], sv[0]);
    if (pass == 1 && c == 0) atomicAdd(&stats[0], sc[0]);
  }
}

template <int C, int D3>
__global__ void k_bn_apply(const float* __restrict__ y, const float* __restrict__ m,
                           const float* __restrict__ stats,
                           const float* __restrict__ gamma,
                           const float* __restrict__ beta,
                           float* __restrict__ out, __bf16* __restrict__ outb,
                           int n) {
  int i = blockIdx.x * blockDim.x + threadIdx.x;
  if (i >= n) return;
  constexpr int CD3 = C * D3;
  int b = (i >= CD3) ? 1 : 0;
  int r = i - b * CD3;
  int c = r / D3;  int sp = r - c * D3;
  float cnt  = stats[0];
  float mean = stats[1 + c] / cnt;
  float var  = stats[1 + C + c] / cnt;
  float scale = gamma[c] * rsqrtf(var + BN_EPS);
  float mv = m[b * D3 + sp];
  float v = mv * fmaxf((y[i] - mean) * scale + beta[c], 0.0f);
  out[i] = v;
  outb[(b * D3 + sp) * C + c] = f2bf(v);
}

template <int C, int D3>
__global__ void k_occ_head(const float* __restrict__ x, const float* __restrict__ m,
                           const float* __restrict__ w, const float* __restrict__ b0,
                           float* __restrict__ p, int n) {
  int i = blockIdx.x * blockDim.x + threadIdx.x;
  if (i >= n) return;
  int b = (i >= D3) ? 1 : 0;
  int sp = i - b * D3;
  float acc = b0[0];
  const float* xb = x + (size_t)(b * C) * D3 + sp;
#pragma unroll 4
  for (int c = 0; c < C; ++c) acc += xb[(size_t)c * D3] * w[c];
  p[i] = m[i] * sigmoidf(acc);
}

template <int C, int D3>
__global__ void k_add_bf(float* __restrict__ a, const float* __restrict__ b,
                         __bf16* __restrict__ ab, int n) {
  int i = blockIdx.x * blockDim.x + threadIdx.x;
  if (i >= n) return;
  constexpr int CD3 = C * D3;
  int bb = (i >= CD3) ? 1 : 0;
  int r = i - bb * CD3;
  int c = r / D3;   int sp = r - c * D3;
  float v = a[i] + b[i];
  a[i] = v;
  ab[(bb * D3 + sp) * C + c] = f2bf(v);
}

template <int C, int D3>
__global__ void k_sigmoid_mask(const float* __restrict__ y,
                               const float* __restrict__ m,
                               float* __restrict__ out, int n) {
  int i = blockIdx.x * blockDim.x + threadIdx.x;
  if (i >= n) return;
  constexpr int CD3 = C * D3;
  int b = (i >= CD3) ? 1 : 0;
  int r = i - b * CD3;
  int c = r / D3;
  int sp = r - c * D3;
  out[i] = sigmoidf(y[i]) * m[b * D3 + sp];
}

// ---------------------------------------------------------------------------
// Host orchestration (graph-capture safe)
// ---------------------------------------------------------------------------
extern "C" void kernel_launch(void* const* d_in, const int* in_sizes, int n_in,
                              void* d_out, int out_size, void* d_ws, size_t ws_size,
                              hipStream_t stream) {
  (void)in_sizes; (void)n_in; (void)out_size; (void)ws_size;

  const float*         feats = (const float*)d_in[0];
  const unsigned char* maskb = (const unsigned char*)d_in[1];
  const float* W1  = (const float*)d_in[2];
  const float* g1  = (const float*)d_in[3];
  const float* b1  = (const float*)d_in[4];
  const float* W2  = (const float*)d_in[5];
  const float* g2  = (const float*)d_in[6];
  const float* b2  = (const float*)d_in[7];
  const float* W3  = (const float*)d_in[8];
  const float* g3  = (const float*)d_in[9];
  const float* b3  = (const float*)d_in[10];
  const float* W4  = (const float*)d_in[11];
  const float* g4  = (const float*)d_in[12];
  const float* b4  = (const float*)d_in[13];
  const float* Wt4 = (const float*)d_in[14];
  const float* gd4 = (const float*)d_in[15];
  const float* bd4 = (const float*)d_in[16];
  const float* wo4 = (const float*)d_in[17];
  const float* bo4 = (const float*)d_in[18];
  const float* Wt3 = (const float*)d_in[19];
  const float* gd3 = (const float*)d_in[20];
  const float* bd3 = (const float*)d_in[21];
  const float* wo3 = (const float*)d_in[22];
  const float* bo3 = (const float*)d_in[23];
  const float* Wt2 = (const float*)d_in[24];
  const float* gd2 = (const float*)d_in[25];
  const float* bd2 = (const float*)d_in[26];
  const float* wo2 = (const float*)d_in[27];
  const float* bo2 = (const float*)d_in[28];
  const float* Wt1 = (const float*)d_in[29];
  const float* wo1 = (const float*)d_in[30];
  const float* bo1 = (const float*)d_in[31];

  const int B = 2;
  const int V96 = 96 * 96 * 96, V48 = 48 * 48 * 48,
            V24 = 24 * 24 * 24, V12 = 12 * 12 * 12, V6 = 6 * 6 * 6;

  // output slices (d0, p4, p3, p2, p1)
  float* out = (float*)d_out;
  float* o_d0 = out;
  float* o_p4 = o_d0 + (size_t)B * 3 * V96;
  float* o_p3 = o_p4 + (size_t)B * V12;
  float* o_p2 = o_p3 + (size_t)B * V24;
  float* o_p1 = o_p2 + (size_t)B * V48;

  // workspace (256B-aligned bump allocator)
  char* wsb = (char*)d_ws;
  size_t off = 0;
  auto alloc = [&](size_t bytes) {
    void* p = wsb + off;
    off += (bytes + 255) & ~(size_t)255;
    return p;
  };
  // bf16 buffers carry a 32B zero guard page before element 0
  auto alloc_bf = [&](size_t elems) {
    __bf16* base = (__bf16*)alloc(elems * 2 + 2 * ZPAD);
    (void)hipMemsetAsync(base, 0, 2 * ZPAD, stream);
    return base + ZPAD;
  };
  float* m0 = (float*)alloc((size_t)B * V96 * 4);
  float* m1 = (float*)alloc((size_t)B * V48 * 4);
  float* m2 = (float*)alloc((size_t)B * V24 * 4);
  float* m3 = (float*)alloc((size_t)B * V12 * 4);
  float* m4 = (float*)alloc((size_t)B * V6 * 4);
  __bf16* xbf  = alloc_bf((size_t)B * 4 * V96);            // Cin padded to 4
  float*  e0   = (float*)alloc((size_t)B * 16 * V48 * 4);
  float*  e1   = (float*)alloc((size_t)B * 32 * V24 * 4);
  float*  e2   = (float*)alloc((size_t)B * 64 * V12 * 4);
  float*  e3   = (float*)alloc((size_t)B * 128 * V6 * 4);
  __bf16* e0bf = alloc_bf((size_t)B * 16 * V48);
  __bf16* e1bf = alloc_bf((size_t)B * 32 * V24);
  __bf16* e2bf = alloc_bf((size_t)B * 64 * V12);
  __bf16* e3bf = alloc_bf((size_t)B * 128 * V6);
  float*  d3   = (float*)alloc((size_t)B * 64 * V12 * 4);
  float*  d2   = (float*)alloc((size_t)B * 32 * V24 * 4);
  float*  d1   = (float*)alloc((size_t)B * 16 * V48 * 4);
  __bf16* d3bf = alloc_bf((size_t)B * 64 * V12);
  __bf16* d2bf = alloc_bf((size_t)B * 32 * V24);
  __bf16* d1bf = alloc_bf((size_t)B * 16 * V48);
  float*  yb   = (float*)alloc((size_t)B * 3 * V96 * 4);   // pre-BN scratch
  __bf16* aswz = (__bf16*)alloc((size_t)221184 * 2);       // max (KP/32)*MT*512
  float*  stats = (float*)alloc((1 + 2 * 128) * 4);

  const int T = 256;
  auto cdiv = [](int a, int b) { return (a + b - 1) / b; };

#define GEMM(CINP_, CINW_, COUT_, DIN_, MODE_, XB_, W_, YP_)                      \
  do {                                                                            \
    constexpr int DOUT_ = (MODE_ == 0) ? (DIN_ / 2) : (DIN_ * 2);                 \
    constexpr int NT_ = 2 * DOUT_ * DOUT_ * DOUT_ / 16;                           \
    constexpr int KP_ = (CINP_ * 27 + 31) & ~31;                                  \
    constexpr int MT_ = ((COUT_ + 15) & ~15) / 16;                                \
    constexpr int TILES_ = (CINP_ == 4) ? NT_ : (NT_ + 1) / 2;                    \
    int na_ = (KP_ / 32) * MT_ * 512;                                             \
    k_pack_a<CINP_, CINW_, COUT_, MODE_>                                          \
        <<<(na_ + 255) / 256, 256, 0, stream>>>(W_, aswz, na_);                   \
    k_gemm_wmma<CINP_, COUT_, DIN_, MODE_>                                        \
        <<<dim3((TILES_ + 7) / 8, MT_), 256, 0, stream>>>(XB_, aswz, YP_);        \
  } while (0)

#define BN(C_, D_, Yp_, mp_, gp_, bp_, outp_, outbp_)                             \
  do {                                                                            \
    constexpr int D3_ = D_ * D_ * D_;                                             \
    (void)hipMemsetAsync(stats, 0, (1 + 2 * 128) * sizeof(float), stream);        \
    int gy_ = (2 * D3_ + 256 * 16 - 1) / (256 * 16);                              \
    gy_ = gy_ < 1 ? 1 : (gy_ > 128 ? 128 : gy_);                                  \
    dim3 gs_(C_, gy_);                                                            \
    k_bn_stats<C_, D3_><<<gs_, 256, 0, stream>>>(Yp_, mp_, stats, 1);             \
    k_bn_stats<C_, D3_><<<gs_, 256, 0, stream>>>(Yp_, mp_, stats, 2);             \
    int n_ = 2 * C_ * D3_;                                                        \
    k_bn_apply<C_, D3_><<<(n_ + 255) / 256, 256, 0, stream>>>(                    \
        Yp_, mp_, stats, gp_, bp_, outp_, outbp_, n_);                            \
  } while (0)

#define HEAD(C_, D_, Xp_, mp_, wp_, bp_, pp_)                                     \
  do {                                                                            \
    constexpr int D3_ = D_ * D_ * D_;                                             \
    int n_ = 2 * D3_;                                                             \
    k_occ_head<C_, D3_><<<(n_ + 255) / 256, 256, 0, stream>>>(Xp_, mp_, wp_,      \
                                                              bp_, pp_, n_);      \
  } while (0)

#define ADDBF(C_, D3_, a_, b_, ab_)                                               \
  do {                                                                            \
    int n_ = 2 * C_ * D3_;                                                        \
    k_add_bf<C_, D3_><<<(n_ + 255) / 256, 256, 0, stream>>>(a_, b_, ab_, n_);     \
  } while (0)

  // ---- masks + masked input (Cin padded to 4, NDHWC bf16) ----
  k_mask_cast<<<cdiv(B * V96, T), T, 0, stream>>>(maskb, m0, B * V96);
  k_apply_mask_c4<V96><<<cdiv(B * V96, T), T, 0, stream>>>(feats, m0, xbf, B * V96);
  k_pool2<48><<<cdiv(B * V48, T), T, 0, stream>>>(m0, m1, B * V48);
  k_pool2<24><<<cdiv(B * V24, T), T, 0, stream>>>(m1, m2, B * V24);
  k_pool2<12><<<cdiv(B * V12, T), T, 0, stream>>>(m2, m3, B * V12);
  k_pool2<6><<<cdiv(B * V6, T), T, 0, stream>>>(m3, m4, B * V6);

  // ---- encoder ----
  GEMM(4, 3, 16, 96, 0, xbf, W1, yb);    BN(16, 48, yb, m1, g1, b1, e0, e0bf);
  GEMM(16, 16, 32, 48, 0, e0bf, W2, yb); BN(32, 24, yb, m2, g2, b2, e1, e1bf);
  GEMM(32, 32, 64, 24, 0, e1bf, W3, yb); BN(64, 12, yb, m3, g3, b3, e2, e2bf);
  GEMM(64, 64, 128, 12, 0, e2bf, W4, yb); BN(128, 6, yb, m4, g4, b4, e3, e3bf);

  // ---- decoder ----
  GEMM(128, 128, 64, 6, 1, e3bf, Wt4, yb); BN(64, 12, yb, m3, gd4, bd4, d3, d3bf);
  HEAD(64, 12, d3, m3, wo4, bo4, o_p4);
  ADDBF(64, V12, d3, e2, d3bf);

  GEMM(64, 64, 32, 12, 1, d3bf, Wt3, yb); BN(32, 24, yb, m2, gd3, bd3, d2, d2bf);
  HEAD(32, 24, d2, m2, wo3, bo3, o_p3);
  ADDBF(32, V24, d2, e1, d2bf);

  GEMM(32, 32, 16, 24, 1, d2bf, Wt2, yb); BN(16, 48, yb, m1, gd2, bd2, d1, d1bf);
  HEAD(16, 48, d1, m1, wo2, bo2, o_p2);
  ADDBF(16, V48, d1, e0, d1bf);

  GEMM(16, 16, 3, 48, 1, d1bf, Wt1, yb);
  k_sigmoid_mask<3, V96><<<cdiv(B * 3 * V96, T), T, 0, stream>>>(yb, m0, o_d0,
                                                                 B * 3 * V96);
  HEAD(3, 96, o_d0, m0, wo1, bo1, o_p1);

#undef GEMM
#undef BN
#undef HEAD
#undef ADDBF
}
